// DTFormer_80745385164967
// MI455X (gfx1250) — compile-verified, matrix operand order checked by reading, not run on hardware
//
#include <hip/hip_runtime.h>
#include <hip/hip_bf16.h>

typedef __attribute__((ext_vector_type(16))) _Float16 v16h;
typedef __attribute__((ext_vector_type(8)))  float    v8f;
typedef __attribute__((ext_vector_type(4)))  float    f32x4;

#define Bb 64
#define Ll 1024
#define Dd 128
#define NKEY 16000      // id in [0,2000) x snap in [1,8] -> key = id*8 + (snap-1)
#define GSTRIDE 136     // padded halves per gtile row (272B = 16B-aligned, bank-skewed)

union AF { v16h v; f32x4 f[2]; };
union BF { v16h v; _Float16 h[16]; };

struct SmemB {
  float     yls[2 * Ll * 2];        // [side][elem][e]  16384 B
  _Float16  gtile[64 * GSTRIDE];    //                  17408 B
  float     aggw1[8 * 132];         // bank-padded       4224 B
  float     aggb1[128];
  float     aggw2[256];
  float     aggb2[2];
  float     encw1[128];
  float     encb1[128];
};
union Smem {                         // hist phase finishes before SmemB is used
  int   hist[NKEY];                  // 64000 B -> union total < 64 KB
  SmemB b;
};

__launch_bounds__(256)
__global__ void dtformer_fused(const int* __restrict__ src_ids,
                               const int* __restrict__ dst_ids,
                               const int* __restrict__ src_sn,
                               const int* __restrict__ dst_sn,
                               const float* __restrict__ agg_w1,
                               const float* __restrict__ agg_b1,
                               const float* __restrict__ agg_w2,
                               const float* __restrict__ agg_b2,
                               const float* __restrict__ enc_w1,
                               const float* __restrict__ enc_b1,
                               const float* __restrict__ enc_w2,
                               const float* __restrict__ enc_b2,
                               float* __restrict__ out)
{
  __shared__ __align__(16) Smem sm;
  const int t    = threadIdx.x;
  const int blk  = blockIdx.x;
  const int lane = t & 31;
  const int wid  = t >> 5;

  const int* sI = src_ids + blk * Ll;
  const int* dI = dst_ids + blk * Ll;
  const int* sS = src_sn  + blk * Ll;
  const int* dS = dst_sn  + blk * Ll;

  // ---- keys (4 src + 4 dst elements per thread) ----
  int ks[4], kd[4];
  #pragma unroll
  for (int j = 0; j < 4; ++j) {
    int i = j * 256 + t;
    ks[j] = sI[i] * 8 + (sS[i] - 1);
    kd[j] = dI[i] * 8 + (dS[i] - 1);
  }

  // ---- pass 1: histogram of src keys ----
  for (int i = t; i < NKEY; i += 256) sm.hist[i] = 0;
  __syncthreads();
  #pragma unroll
  for (int j = 0; j < 4; ++j)
    if ((unsigned)ks[j] < (unsigned)NKEY) atomicAdd(&sm.hist[ks[j]], 1);
  __syncthreads();
  int s_self[4], d_cross[4];
  #pragma unroll
  for (int j = 0; j < 4; ++j) {
    s_self[j]  = ((unsigned)ks[j] < (unsigned)NKEY) ? sm.hist[ks[j]] : 0;
    d_cross[j] = ((unsigned)kd[j] < (unsigned)NKEY) ? sm.hist[kd[j]] : 0;
  }
  __syncthreads();

  // ---- pass 2: histogram of dst keys ----
  for (int i = t; i < NKEY; i += 256) sm.hist[i] = 0;
  __syncthreads();
  #pragma unroll
  for (int j = 0; j < 4; ++j)
    if ((unsigned)kd[j] < (unsigned)NKEY) atomicAdd(&sm.hist[kd[j]], 1);
  __syncthreads();
  int s_cross[4], d_self[4];
  #pragma unroll
  for (int j = 0; j < 4; ++j) {
    s_cross[j] = ((unsigned)ks[j] < (unsigned)NKEY) ? sm.hist[ks[j]] : 0;
    d_self[j]  = ((unsigned)kd[j] < (unsigned)NKEY) ? sm.hist[kd[j]] : 0;
  }
  __syncthreads();   // hist region dead from here; SmemB overlays it

  // ---- stage small weights to LDS ----
  for (int i = t; i < 8 * 128; i += 256)
    sm.b.aggw1[(i >> 7) * 132 + (i & 127)] = agg_w1[i];
  for (int i = t; i < 128; i += 256) {
    sm.b.aggb1[i] = agg_b1[i];
    sm.b.encw1[i] = enc_w1[i];
    sm.b.encb1[i] = enc_b1[i];
  }
  for (int i = t; i < 256; i += 256) sm.b.aggw2[i] = agg_w2[i];
  if (t < 2) sm.b.aggb2[t] = agg_b2[t];
  __syncthreads();

  // ---- agg MLP: per-element snapshot features y[side][i][0..1] ----
  // one-hot over snapshots => only row s* of agg_w1; mean over channels is
  // symmetric, so the dst channel flip is a no-op.
  #pragma unroll
  for (int j = 0; j < 4; ++j) {
    int i = j * 256 + t;
    #pragma unroll
    for (int side = 0; side < 2; ++side) {
      int key  = side ? kd[j] : ks[j];
      float cs = (float)(side ? d_self[j]  : s_self[j]);
      float cc = (float)(side ? d_cross[j] : s_cross[j]);
      float v  = ((key >> 3) != 0) ? 1.f : 0.f;   // id != 0
      float xs = cs * v, xc = cc * v;
      const float* w1row = &sm.b.aggw1[(key & 7) * 132];
      float a0 = 0.f, a1 = 0.f;
      for (int d = 0; d < 128; ++d) {
        float w = w1row[d], bb = sm.b.aggb1[d];
        float h = fmaxf(xs * w + bb, 0.f) + fmaxf(xc * w + bb, 0.f);
        a0 += h * sm.b.aggw2[2 * d];
        a1 += h * sm.b.aggw2[2 * d + 1];
      }
      sm.b.yls[(side * Ll + i) * 2 + 0] = 0.5f * a0 + sm.b.aggb2[0];
      sm.b.yls[(side * Ll + i) * 2 + 1] = 0.5f * a1 + sm.b.aggb2[1];
    }
  }
  __syncthreads();

  // ---- encode GEMM: out = G @ enc_w2 + 2*enc_b2 via v_wmma_f32_16x16x32_f16
  // wave 'wid' owns output columns n0..n0+15; B fragments live in registers.
  // 16-bit fragment layout: lanes 0-15 hold K {0..7,16..23}, lanes 16-31
  // hold K {8..15,24..31} (two packed f16 per VGPR).
  const int n0    = wid * 16;
  const int col   = n0 + (lane & 15);
  const int khalf = (lane >> 4) * 8;   // 0 or 8
  BF bf[4];
  #pragma unroll
  for (int kk = 0; kk < 4; ++kk) {
    int kl = kk * 32 + khalf;
    #pragma unroll
    for (int jj = 0; jj < 8; ++jj)
      bf[kk].h[jj]     = (_Float16)enc_w2[(kl + jj) * 128 + col];
    #pragma unroll
    for (int jj = 0; jj < 8; ++jj)
      bf[kk].h[8 + jj] = (_Float16)enc_w2[(kl + 16 + jj) * 128 + col];
  }
  const float bias2 = 2.f * enc_b2[col];

  for (int side = 0; side < 2; ++side) {
    float* outS = out + (size_t)side * Bb * Ll * Dd + (size_t)blk * Ll * Dd;
    for (int ch = 0; ch < 16; ++ch) {
      const int row0 = ch * 64;
      // cooperatively build 64 rows of G = sum_c relu(y_c*enc_w1 + enc_b1)
      #pragma unroll 4
      for (int it = 0; it < 32; ++it) {
        int flat = it * 256 + t;
        int r = flat >> 7, k = flat & 127;
        float y0 = sm.b.yls[(side * Ll + row0 + r) * 2 + 0];
        float y1 = sm.b.yls[(side * Ll + row0 + r) * 2 + 1];
        float w = sm.b.encw1[k], bb = sm.b.encb1[k];
        float g = fmaxf(y0 * w + bb, 0.f) + fmaxf(y1 * w + bb, 0.f);
        sm.b.gtile[r * GSTRIDE + k] = (_Float16)g;
      }
      __syncthreads();
      #pragma unroll
      for (int mt = 0; mt < 4; ++mt) {
        const _Float16* gp = &sm.b.gtile[(mt * 16 + (lane & 15)) * GSTRIDE];
        v8f c = {0.f, 0.f, 0.f, 0.f, 0.f, 0.f, 0.f, 0.f};
        #pragma unroll
        for (int kk = 0; kk < 4; ++kk) {
          AF a;
          a.f[0] = *(const f32x4*)(gp + kk * 32 + khalf);        // ds_load_b128
          a.f[1] = *(const f32x4*)(gp + kk * 32 + khalf + 16);   // ds_load_b128
          c = __builtin_amdgcn_wmma_f32_16x16x32_f16(
                false, a.v, false, bf[kk].v, (short)0, c, false, false);
        }
        // C layout: VGPR j -> M=j (lanes 0-15) / M=j+8 (lanes 16-31)
        const int rbase = row0 + mt * 16 + khalf;
        #pragma unroll
        for (int jj = 0; jj < 8; ++jj)
          outS[(size_t)(rbase + jj) * Dd + col] = c[jj] + bias2;
      }
      __syncthreads();
    }
  }
}

extern "C" void kernel_launch(void* const* d_in, const int* in_sizes, int n_in,
                              void* d_out, int out_size, void* d_ws, size_t ws_size,
                              hipStream_t stream) {
  (void)in_sizes; (void)n_in; (void)out_size; (void)d_ws; (void)ws_size;
  const int*   src_ids = (const int*)d_in[0];
  const int*   dst_ids = (const int*)d_in[1];
  const int*   src_sn  = (const int*)d_in[2];
  const int*   dst_sn  = (const int*)d_in[3];
  // d_in[4] = num_snapshots (compile-time S=8)
  const float* agg_w1  = (const float*)d_in[5];
  const float* agg_b1  = (const float*)d_in[6];
  const float* agg_w2  = (const float*)d_in[7];
  const float* agg_b2  = (const float*)d_in[8];
  const float* enc_w1  = (const float*)d_in[9];
  const float* enc_b1  = (const float*)d_in[10];
  const float* enc_w2  = (const float*)d_in[11];
  const float* enc_b2  = (const float*)d_in[12];
  float* outp = (float*)d_out;

  dtformer_fused<<<Bb, 256, 0, stream>>>(src_ids, dst_ids, src_sn, dst_sn,
                                         agg_w1, agg_b1, agg_w2, agg_b2,
                                         enc_w1, enc_b1, enc_w2, enc_b2, outp);
}
